// DeepLSTM_52080773431929
// MI455X (gfx1250) — matrix operand order repaired and checked by reading, not compiled, not run on hardware
//
#include <hip/hip_runtime.h>

// ---------------------------------------------------------------------------
// Persistent wavefront-pipelined deep LSTM for MI455X (gfx1250, wave32).
//   B=32, T=512, I=256, H=512, L=4.
// GEMMs use v_wmma_f32_16x16x32_f16. Weights pre-swizzled to WMMA B-fragment
// layout (f16, 15.7MB -> L2 resident). Activations staged into LDS once per
// phase via GLOBAL_LOAD_ASYNC_TO_LDS_B128 (ASYNCcnt), then read as WMMA A
// fragments with ds_load_b128 (bank-conflict-free via +8h row pad).
// 64 persistent WGs (4 layers x 16 column slices), producer/consumer sync via
// agent-scope acquire/release atomics + s_sleep backoff.
// ---------------------------------------------------------------------------

typedef __attribute__((ext_vector_type(16))) _Float16 v16h;
typedef __attribute__((ext_vector_type(8)))  float    v8f;

#define LSTM_B 32
#define LSTM_T 512
#define LSTM_I 256
#define LSTM_H 512
#define LSTM_L 4
#define NSLICE 16            // column slices per layer (32 cols each)
#define APAD   520           // A-stage row stride in halves (512 + 8 pad)

// packed weight sizes in halves: [ntile(128)][kblock][lane(32)][16 halves]
#define PW_L0   (128 * 24 * 512)           // layer 0: K = 256+512 -> 24 kblocks
#define PW_LX   (128 * 32 * 512)           // layers 1..3: K = 1024 -> 32 kblocks
#define PW_TOT  (PW_L0 + 3 * PW_LX)        // 7,864,320 halves
#define XF_TOT  ((size_t)LSTM_T * LSTM_B * LSTM_I)                 // 4,194,304
#define HALL_TOT ((size_t)LSTM_L * LSTM_T * LSTM_B * LSTM_H)       // 33,554,432

union FragH { v16h v; uint4 q[2]; };

__device__ __forceinline__ float sigf(float x)   { return 1.0f / (1.0f + __expf(-x)); }
__device__ __forceinline__ float tanhf_(float x) { return 2.0f / (1.0f + __expf(-2.0f * x)) - 1.0f; }

// ---------------------------------------------------------------------------
// Pack kernel: x -> f16 time-major; weights -> f16 WMMA B-fragment layout.
// B 32x16 f16 fragment: lane holds col n = ntile*16 + (lane&15),
// k = kb*32 + (lane>>4)*16 + e, e = 0..15 contiguous halves.
// Gate-major columns: col = gate*512 + j (gates i,f,o,c).
// Combined K rows: [Wx rows (Kx), then Wh rows (512)].
// ---------------------------------------------------------------------------
__global__ void lstm_pack_kernel(
    const float* __restrict__ inputs,
    const float* __restrict__ Wxi0, const float* __restrict__ Wxf0,
    const float* __restrict__ Wxo0, const float* __restrict__ Wxc0,
    const float* __restrict__ Wxi,  const float* __restrict__ Wxf,
    const float* __restrict__ Wxo,  const float* __restrict__ Wxc,
    const float* __restrict__ Whi,  const float* __restrict__ Whf,
    const float* __restrict__ Who,  const float* __restrict__ Whc,
    _Float16* __restrict__ xf16, _Float16* __restrict__ packW)
{
  const size_t stride = (size_t)gridDim.x * blockDim.x;
  // x conversion: [B,T,I] f32 -> [T,B,I] f16
  for (size_t q = (size_t)blockIdx.x * blockDim.x + threadIdx.x; q < XF_TOT; q += stride) {
    int i = (int)(q % LSTM_I);
    int b = (int)((q / LSTM_I) % LSTM_B);
    int t = (int)(q / ((size_t)LSTM_I * LSTM_B));
    xf16[((size_t)t * LSTM_B + b) * LSTM_I + i] =
        (_Float16)inputs[((size_t)b * LSTM_T + t) * LSTM_I + i];
  }
  // weight fragment packing
  for (size_t q = (size_t)blockIdx.x * blockDim.x + threadIdx.x; q < (size_t)PW_TOT; q += stride) {
    int l; size_t rem;
    if (q < (size_t)PW_L0) { l = 0; rem = q; }
    else { l = 1 + (int)((q - PW_L0) / PW_LX); rem = (q - PW_L0) % PW_LX; }
    const int KB = (l == 0) ? 24 : 32;
    int e    = (int)(rem & 15);
    int lane = (int)((rem >> 4) & 31);
    size_t t2 = rem >> 9;                 // = ntile*KB + kb
    int kb = (int)(t2 % KB);
    int nt = (int)(t2 / KB);
    int k   = kb * 32 + ((lane >> 4) << 4) + e;
    int col = nt * 16 + (lane & 15);
    int g = col >> 9;
    int j = col & 511;
    int Kx = (l == 0) ? LSTM_I : LSTM_H;
    float v;
    if (k < Kx) {
      if (l == 0) {
        const float* w = (g == 0) ? Wxi0 : (g == 1) ? Wxf0 : (g == 2) ? Wxo0 : Wxc0;
        v = w[(size_t)k * LSTM_H + j];
      } else {
        const float* w = (g == 0) ? Wxi : (g == 1) ? Wxf : (g == 2) ? Wxo : Wxc;
        v = w[((size_t)(l - 1) * LSTM_H + k) * LSTM_H + j];
      }
    } else {
      int kh = k - Kx;
      const float* w = (g == 0) ? Whi : (g == 1) ? Whf : (g == 2) ? Who : Whc;
      v = w[((size_t)l * LSTM_H + kh) * LSTM_H + j];
    }
    packW[q] = (_Float16)v;
  }
}

// ---------------------------------------------------------------------------
// Persistent recurrent kernel. Grid = 64 blocks (layer = bx>>4, slice = bx&15),
// 256 threads = 8 waves. Wave w: gate g = w>>1, tile pair p = w&1,
// N-tile nt = g*32 + s*2 + p (16 gate columns). Each wave computes both
// M-tiles (rows 0-15, 16-31) sharing the B fragment -> 2 WMMA per k-step.
// ---------------------------------------------------------------------------
__global__ __launch_bounds__(256, 1) void lstm_persist_kernel(
    const _Float16* __restrict__ xf16, const _Float16* __restrict__ packW,
    _Float16* __restrict__ h_all, int* __restrict__ flags,
    const float* __restrict__ bi, const float* __restrict__ bf,
    const float* __restrict__ bo, const float* __restrict__ bc,
    float* __restrict__ out)
{
  const int l   = blockIdx.x >> 4;
  const int s   = blockIdx.x & 15;
  const int tid = threadIdx.x;
  const int w    = tid >> 5;
  const int lane = tid & 31;
  const int g = w >> 1;
  const int p = w & 1;
  const int nt  = g * 32 + s * 2 + p;
  const int KBx = (l == 0) ? 8 : 16;          // x-part k-blocks (Kx/32)
  const int KB  = KBx + 16;                   // + h-part (512/32)
  const size_t woff = (l == 0) ? 0 : (size_t)PW_L0 + (size_t)(l - 1) * PW_LX;
  const _Float16* wbase = packW + woff + ((size_t)nt * KB) * 512 + (size_t)lane * 16;

  const int hi = lane >> 4;     // A-frag k-subgroup, C-row group
  const int m0 = lane & 15;     // A-frag row (M-tile 0)
  const int n  = lane & 15;     // C column within tile

  __shared__ __align__(16) _Float16 aS[LSTM_B][APAD];   // staged activations
  __shared__ float gatesS[4][LSTM_B][32];
  __shared__ float cbuf[LSTM_B][32];
  for (int idx = tid; idx < LSTM_B * 32; idx += 256) cbuf[idx >> 5][idx & 31] = 0.0f;
  __syncthreads();

  // hoist per-thread bias values (constant over t)
  float bir[4], bfr_[4], bor[4], bcr[4];
  #pragma unroll
  for (int r = 0; r < 4; ++r) {
    int jg = s * 32 + ((tid + r * 256) & 31);
    bir[r]  = bi[l * LSTM_H + jg];
    bfr_[r] = bf[l * LSTM_H + jg];
    bor[r]  = bo[l * LSTM_H + jg];
    bcr[r]  = bc[l * LSTM_H + jg];
  }

  for (int t = 0; t < LSTM_T; ++t) {
    // ---- wait for producers (acquire invalidates WGP$ for the whole WG) ----
    if (tid == 0) {
      if (l > 0) {
        int* f = &flags[(l - 1) * LSTM_T + t];
        while (__hip_atomic_load(f, __ATOMIC_ACQUIRE, __HIP_MEMORY_SCOPE_AGENT) < NSLICE)
          __builtin_amdgcn_s_sleep(2);
      }
      if (t > 0) {
        int* f = &flags[l * LSTM_T + (t - 1)];
        while (__hip_atomic_load(f, __ATOMIC_ACQUIRE, __HIP_MEMORY_SCOPE_AGENT) < NSLICE)
          __builtin_amdgcn_s_sleep(2);
      }
    }
    __syncthreads();

    v8f acc0 = {0.f,0.f,0.f,0.f,0.f,0.f,0.f,0.f};
    v8f acc1 = {0.f,0.f,0.f,0.f,0.f,0.f,0.f,0.f};
    const _Float16* wp = wbase;

    // ================= phase X: input part (x_t or h_{l-1,t}) =============
    {
      const _Float16* asrc; int rs, sh;
      if (l == 0) { asrc = xf16 + (size_t)t * LSTM_B * LSTM_I; rs = LSTM_I; sh = 5; }
      else        { asrc = h_all + ((size_t)((l - 1) * LSTM_T + t) * LSTM_B) * LSTM_H;
                    rs = LSTM_H; sh = 6; }
      // async-stage 32 x Kx halves into LDS (16B per lane per issue)
      const int nchunk = LSTM_B << sh;               // 32 * (Kx/8)
      for (int c = tid; c < nchunk; c += 256) {
        int b  = c >> sh;
        int ko = (c & ((1 << sh) - 1)) << 3;
        unsigned ldsa = (unsigned)(size_t)&aS[b][ko];
        const _Float16* gp = asrc + (size_t)b * rs + ko;
        asm volatile("global_load_async_to_lds_b128 %0, %1, off"
                     :: "v"(ldsa), "v"(gp) : "memory");
      }
      asm volatile("s_wait_asynccnt 0x0" ::: "memory");
      __syncthreads();

      #pragma unroll 4
      for (int kb = 0; kb < KBx; ++kb) {
        const int kofs = kb * 32 + hi * 16;
        FragH a0, a1, bfrag;
        const _Float16* ap0 = &aS[m0][kofs];
        a0.q[0] = *(const uint4*)(ap0);
        a0.q[1] = *(const uint4*)(ap0 + 8);
        const _Float16* ap1 = &aS[m0 + 16][kofs];
        a1.q[0] = *(const uint4*)(ap1);
        a1.q[1] = *(const uint4*)(ap1 + 8);
        bfrag.q[0] = *(const uint4*)(wp);
        bfrag.q[1] = *(const uint4*)(wp + 8);
        acc0 = __builtin_amdgcn_wmma_f32_16x16x32_f16(false, a0.v, false, bfrag.v,
                                                      (short)0, acc0, false, false);
        acc1 = __builtin_amdgcn_wmma_f32_16x16x32_f16(false, a1.v, false, bfrag.v,
                                                      (short)0, acc1, false, false);
        wp += 512;
      }
    }
    __syncthreads();   // aS dead: all waves finished phase-X reads

    // ================= phase H: recurrent part h_{l,t-1} (skip t=0) ========
    if (t > 0) {
      const _Float16* asrc = h_all + ((size_t)(l * LSTM_T + (t - 1)) * LSTM_B) * LSTM_H;
      for (int c = tid; c < (LSTM_B << 6); c += 256) {
        int b  = c >> 6;
        int ko = (c & 63) << 3;
        unsigned ldsa = (unsigned)(size_t)&aS[b][ko];
        const _Float16* gp = asrc + (size_t)b * LSTM_H + ko;
        asm volatile("global_load_async_to_lds_b128 %0, %1, off"
                     :: "v"(ldsa), "v"(gp) : "memory");
      }
      asm volatile("s_wait_asynccnt 0x0" ::: "memory");
      __syncthreads();

      #pragma unroll 4
      for (int kb = 0; kb < 16; ++kb) {
        const int kofs = kb * 32 + hi * 16;
        FragH a0, a1, bfrag;
        const _Float16* ap0 = &aS[m0][kofs];
        a0.q[0] = *(const uint4*)(ap0);
        a0.q[1] = *(const uint4*)(ap0 + 8);
        const _Float16* ap1 = &aS[m0 + 16][kofs];
        a1.q[0] = *(const uint4*)(ap1);
        a1.q[1] = *(const uint4*)(ap1 + 8);
        bfrag.q[0] = *(const uint4*)(wp);
        bfrag.q[1] = *(const uint4*)(wp + 8);
        acc0 = __builtin_amdgcn_wmma_f32_16x16x32_f16(false, a0.v, false, bfrag.v,
                                                      (short)0, acc0, false, false);
        acc1 = __builtin_amdgcn_wmma_f32_16x16x32_f16(false, a1.v, false, bfrag.v,
                                                      (short)0, acc1, false, false);
        wp += 512;
      }
    }

    // ---- spill gate pre-activations to LDS (C layout: V[r] = row hi*8+r) ----
    {
      const int mo = hi * 8;
      const int cl = p * 16 + n;
      #pragma unroll
      for (int r = 0; r < 8; ++r) {
        gatesS[g][mo + r][cl]      = acc0[r];
        gatesS[g][16 + mo + r][cl] = acc1[r];
      }
    }
    __syncthreads();

    // ---- elementwise LSTM cell: 1024 (b,j) pairs over 256 threads ----
    const size_t hdst = ((size_t)(l * LSTM_T + t) * LSTM_B) * LSTM_H;
    #pragma unroll
    for (int r = 0; r < 4; ++r) {
      int idx = tid + r * 256;
      int m = idx >> 5, j = idx & 31;
      int jg = s * 32 + j;
      float it = sigf(gatesS[0][m][j] + bir[r]);
      float ft = sigf(gatesS[1][m][j] + bfr_[r]);
      float ot = sigf(gatesS[2][m][j] + bor[r]);
      float ct = tanhf_(gatesS[3][m][j] + bcr[r]);
      float c  = ft * cbuf[m][j] + it * ct;
      cbuf[m][j] = c;
      float h = ot * tanhf_(c);
      h_all[hdst + (size_t)m * LSTM_H + jg] = (_Float16)h;
      if (l == LSTM_L - 1)
        out[((size_t)m * LSTM_T + t) * LSTM_H + jg] = h;
    }
    __threadfence();
    __syncthreads();
    if (tid == 0)
      __hip_atomic_fetch_add(&flags[l * LSTM_T + t], 1,
                             __ATOMIC_RELEASE, __HIP_MEMORY_SCOPE_AGENT);
    // top-of-loop barrier (after tid0 spin) orders everything for next step
  }
}

// ---------------------------------------------------------------------------
extern "C" void kernel_launch(void* const* d_in, const int* in_sizes, int n_in,
                              void* d_out, int out_size, void* d_ws, size_t ws_size,
                              hipStream_t stream)
{
  const float* inputs = (const float*)d_in[0];
  const float* Wxi0 = (const float*)d_in[1];
  const float* Wxf0 = (const float*)d_in[2];
  const float* Wxo0 = (const float*)d_in[3];
  const float* Wxc0 = (const float*)d_in[4];
  const float* Wxi  = (const float*)d_in[5];
  const float* Wxf  = (const float*)d_in[6];
  const float* Wxo  = (const float*)d_in[7];
  const float* Wxc  = (const float*)d_in[8];
  const float* Whi  = (const float*)d_in[9];
  const float* Whf  = (const float*)d_in[10];
  const float* Who  = (const float*)d_in[11];
  const float* Whc  = (const float*)d_in[12];
  const float* bi   = (const float*)d_in[13];
  const float* bf   = (const float*)d_in[14];
  const float* bo   = (const float*)d_in[15];
  const float* bc   = (const float*)d_in[16];
  float* out = (float*)d_out;

  _Float16* xf16  = (_Float16*)d_ws;
  _Float16* packW = xf16 + XF_TOT;
  _Float16* h_all = packW + PW_TOT;
  int* flags = (int*)(h_all + HALL_TOT);

  const size_t needed = (XF_TOT + (size_t)PW_TOT + HALL_TOT) * sizeof(_Float16)
                      + (size_t)LSTM_L * LSTM_T * sizeof(int);
  if (ws_size < needed) return;  // workspace too small; cannot run safely

  hipMemsetAsync(flags, 0, LSTM_L * LSTM_T * sizeof(int), stream);
  lstm_pack_kernel<<<2048, 256, 0, stream>>>(
      inputs, Wxi0, Wxf0, Wxo0, Wxc0, Wxi, Wxf, Wxo, Wxc,
      Whi, Whf, Who, Whc, xf16, packW);
  lstm_persist_kernel<<<LSTM_L * NSLICE, 256, 0, stream>>>(
      xf16, packW, h_all, flags, bi, bf, bo, bc, out);
}
